// LocallyConnectedNN_1056561955144
// MI455X (gfx1250) — matrix operand order repaired and checked by reading, not compiled, not run on hardware
//
#include <hip/hip_runtime.h>

#define BATCH 1024
#define MBLOCK 128   // batch rows staged per workgroup (8 mtiles, 8 waves)

typedef __attribute__((ext_vector_type(16))) __bf16 v16bf;
typedef __attribute__((ext_vector_type(8)))  float  v8f;

union Frag16 { v16bf v; unsigned int u[8]; };
union FragC  { v8f   v; float        f[8]; };

// ---------- bf16 split helpers (value = hi + lo, RNE both halves) ----------
__device__ __forceinline__ unsigned int f2bf(float f) {
  unsigned int u = __float_as_uint(f);
  u += 0x7FFFu + ((u >> 16) & 1u);
  return u >> 16;
}
__device__ __forceinline__ float bf2f(unsigned int b) { return __uint_as_float(b << 16); }
__device__ __forceinline__ unsigned int pack_split(float f) {
  unsigned int hi = f2bf(f);
  unsigned int lo = f2bf(f - bf2f(hi));
  return (hi << 16) | lo;
}

// ---------- prep kernels ----------
__global__ void zero_stats(float* s, int n) {
  int t = blockIdx.x * blockDim.x + threadIdx.x;
  if (t < n) s[t] = 0.0f;
}

// x: (B,1,16,16) f32 -> act0 packed (1,16,16,B)
__global__ void pack_input(const float* __restrict__ x, unsigned int* __restrict__ act0) {
  int t = blockIdx.x * blockDim.x + threadIdx.x;
  if (t >= 256 * BATCH) return;
  int hw = t >> 10, b = t & (BATCH - 1);
  act0[t] = pack_split(x[b * 256 + hw]);
}

// W: (O,Cin,OH,OW,3,3) -> hi/lo bf16 planes in (loc, O, Kpad), zero K-padding
__global__ void pack_lcw(const float* __restrict__ W, unsigned short* __restrict__ whi,
                         unsigned short* __restrict__ wlo,
                         int O, int Cin, int OHW, int Kc, int Kpad) {
  int t = blockIdx.x * blockDim.x + threadIdx.x;
  int total = OHW * O * Kpad;
  if (t >= total) return;
  int loc = t / (O * Kpad);
  int r   = t % (O * Kpad);
  int o = r / Kpad, k = r % Kpad;
  float w = 0.0f;
  if (k < Kc) {
    int c = k / 9, ij = k % 9;
    w = W[(size_t)(o * Cin + c) * OHW * 9 + (size_t)loc * 9 + ij];
  }
  unsigned int hi = f2bf(w);
  whi[t] = (unsigned short)hi;
  wlo[t] = (unsigned short)f2bf(w - bf2f(hi));
}

// fcW: (10,6400) -> hi/lo planes (16,6400), rows 10..15 zero (layouts coincide)
__global__ void pack_fcw(const float* __restrict__ fcW, unsigned short* __restrict__ whi,
                         unsigned short* __restrict__ wlo) {
  int t = blockIdx.x * blockDim.x + threadIdx.x;
  if (t >= 16 * 6400) return;
  float w = (t < 10 * 6400) ? fcW[t] : 0.0f;
  unsigned int hi = f2bf(w);
  whi[t] = (unsigned short)hi;
  wlo[t] = (unsigned short)f2bf(w - bf2f(hi));
}

// ---------- locally-connected layer: per-location WMMA GEMM ----------
// act: packed (Cin,Hin,Win,B); whi/wlo: (loc, O=NT*16, Kpad); bias: (O,OH,OW)
// raw: f32 (O,OH*OW,B); sum/sumsq: per-channel BN stats via atomics.
// Block = 256 threads (8 waves), grid = (OH*OW, B/MBLOCK).
// LDS: hi/lo bf16 planes, batch-major [b_local][k] -> A slot = 1 ds_load_b32.
template <int NT>
__global__ void __launch_bounds__(256)
lc_gemm(const unsigned int* __restrict__ act,
        const unsigned short* __restrict__ whi, const unsigned short* __restrict__ wlo,
        const float* __restrict__ bias, float* __restrict__ raw,
        float* __restrict__ sum, float* __restrict__ sumsq,
        int Cin, int Hin, int Win, int OH, int OW, int Kc, int Kpad) {
  extern __shared__ unsigned short lds[];
  unsigned short* lhi = lds;
  unsigned short* llo = lds + MBLOCK * Kpad;

  const int O     = NT * 16;
  const int loc   = blockIdx.x;
  const int OHW   = OH * OW;
  const int oh    = loc / OW, ow = loc % OW;
  const int bbase = blockIdx.y * MBLOCK;
  const int tid   = threadIdx.x;

  // ---- stage activation patch: b128 global loads, de-interleave + transpose ----
  const int QROW = MBLOCK / 4;              // uint4 chunks per k-row
  for (int idx = tid; idx < Kpad * QROW; idx += 256) {
    int k = idx / QROW, q = idx % QROW;
    uint4 val = {0u, 0u, 0u, 0u};
    if (k < Kc) {
      int c = k / 9, ij = k % 9, ii = ij / 3, jj = ij % 3;
      const uint4* p = (const uint4*)(act +
          ((size_t)(c * Hin + oh + ii) * Win + (ow + jj)) * BATCH + bbase);
      val = p[q];
    }
    int bl = q * 4;
    lhi[(bl + 0) * Kpad + k] = (unsigned short)(val.x >> 16);
    llo[(bl + 0) * Kpad + k] = (unsigned short)(val.x & 0xFFFFu);
    lhi[(bl + 1) * Kpad + k] = (unsigned short)(val.y >> 16);
    llo[(bl + 1) * Kpad + k] = (unsigned short)(val.y & 0xFFFFu);
    lhi[(bl + 2) * Kpad + k] = (unsigned short)(val.z >> 16);
    llo[(bl + 2) * Kpad + k] = (unsigned short)(val.z & 0xFFFFu);
    lhi[(bl + 3) * Kpad + k] = (unsigned short)(val.w >> 16);
    llo[(bl + 3) * Kpad + k] = (unsigned short)(val.w & 0xFFFFu);
  }
  __syncthreads();

  // ---- compute: one wave per 16-batch tile, NT accumulator tiles over N ----
  const int wave  = tid >> 5, lane = tid & 31;
  const int halfL = lane >> 4, l16 = lane & 15;
  const unsigned short* ahp = lhi + (wave * 16 + l16) * Kpad;  // A row (m = l16)
  const unsigned short* alp = llo + (wave * 16 + l16) * Kpad;

  FragC acc[NT];
#pragma unroll
  for (int nt = 0; nt < NT; ++nt)
#pragma unroll
    for (int i = 0; i < 8; ++i) acc[nt].f[i] = 0.0f;

  const size_t wbase = ((size_t)loc * O + l16) * Kpad;  // B col (n = nt*16 + l16)

  for (int kc = 0; kc < Kpad; kc += 32) {
    Frag16 ahi, alo;
#pragma unroll
    for (int v = 0; v < 8; ++v) {
      int kb = kc + ((v >> 2) << 4) + (halfL << 3) + ((v & 3) << 1);
      ahi.u[v] = *(const unsigned int*)(ahp + kb);
      alo.u[v] = *(const unsigned int*)(alp + kb);
    }
#pragma unroll
    for (int nt = 0; nt < NT; ++nt) {
      const unsigned short* bhp = whi + wbase + (size_t)nt * 16 * Kpad;
      const unsigned short* blp = wlo + wbase + (size_t)nt * 16 * Kpad;
      Frag16 bhi, blo;
#pragma unroll
      for (int v = 0; v < 8; ++v) {
        int kB = kc + ((v >> 2) << 4) + (halfL << 3) + ((v & 3) << 1);
        bhi.u[v] = *(const unsigned int*)(bhp + kB);
        blo.u[v] = *(const unsigned int*)(blp + kB);
      }
      // split-bf16 fp32 emulation: AB ~= Ahi*Bhi + Ahi*Blo + Alo*Bhi
      acc[nt].v = __builtin_amdgcn_wmma_f32_16x16x32_bf16(false, ahi.v, false, bhi.v, (short)0, acc[nt].v, false, false);
      acc[nt].v = __builtin_amdgcn_wmma_f32_16x16x32_bf16(false, ahi.v, false, blo.v, (short)0, acc[nt].v, false, false);
      acc[nt].v = __builtin_amdgcn_wmma_f32_16x16x32_bf16(false, alo.v, false, bhi.v, (short)0, acc[nt].v, false, false);
    }
  }

  // ---- epilogue: bias, raw store (contiguous 8 floats/lane), BN stats ----
#pragma unroll
  for (int nt = 0; nt < NT; ++nt) {
    int n = nt * 16 + l16;
    float bv = bias[n * OHW + loc];
    float s = 0.0f, s2 = 0.0f;
    float* rp = raw + ((size_t)n * OHW + loc) * BATCH + bbase + wave * 16 + halfL * 8;
#pragma unroll
    for (int v = 0; v < 8; ++v) {
      float val = acc[nt].f[v] + bv;   // C layout: m = v + 8*halfL
      rp[v] = val;
      s += val; s2 += val * val;
    }
    atomicAdd(&sum[n], s);
    atomicAdd(&sumsq[n], s2);
  }
}

// ---------- BN (batch stats) + activation -> packed split-bf16 (layers 1,2) ----------
__global__ void bn_act(const float* __restrict__ raw, const float* __restrict__ sum,
                       const float* __restrict__ sumsq, const float* __restrict__ g,
                       const float* __restrict__ be, unsigned int* __restrict__ out,
                       int OHW, int total) {
  int t = blockIdx.x * blockDim.x + threadIdx.x;
  if (t >= total) return;
  int per = OHW * BATCH;
  int o = t / per;
  float N = (float)per;
  float mean = sum[o] / N;
  float var  = sumsq[o] / N - mean * mean;
  float y = g[o] * (raw[t] - mean) * rsqrtf(var + 1e-5f) + be[o];
  out[t] = pack_split(fmaxf(y, 0.0f));
}

// ---------- BN + tanh -> hi/lo planes (B,6400) for the FC layer ----------
__global__ void bn_act_fc(const float* __restrict__ raw, const float* __restrict__ sum,
                          const float* __restrict__ sumsq, const float* __restrict__ g,
                          const float* __restrict__ be,
                          unsigned short* __restrict__ ahi, unsigned short* __restrict__ alo) {
  int t = blockIdx.x * blockDim.x + threadIdx.x;
  if (t >= 64 * 100 * BATCH) return;
  int per = 100 * BATCH;
  int o = t / per, rem = t % per;
  int s = rem >> 10, b = rem & (BATCH - 1);
  float N = (float)per;
  float mean = sum[o] / N;
  float var  = sumsq[o] / N - mean * mean;
  float y = tanhf(g[o] * (raw[t] - mean) * rsqrtf(var + 1e-5f) + be[o]);
  unsigned int hi = f2bf(y);
  size_t idx = (size_t)b * 6400 + o * 100 + s;   // h.reshape(B,-1) feature index
  ahi[idx] = (unsigned short)hi;
  alo[idx] = (unsigned short)f2bf(y - bf2f(hi));
}

// ---------- final FC: (1024,6400) @ (6400,10) via WMMA ----------
__global__ void __launch_bounds__(32)
fc_gemm(const unsigned short* __restrict__ ahip, const unsigned short* __restrict__ alop,
        const unsigned short* __restrict__ whi, const unsigned short* __restrict__ wlo,
        const float* __restrict__ fcb, float* __restrict__ out) {
  const int mtile = blockIdx.x;
  const int lane = threadIdx.x;
  const int halfL = lane >> 4, l16 = lane & 15;
  const unsigned short* ah = ahip + (size_t)(mtile * 16 + l16) * 6400;
  const unsigned short* al = alop + (size_t)(mtile * 16 + l16) * 6400;
  const unsigned short* bh = whi + (size_t)l16 * 6400;
  const unsigned short* bl = wlo + (size_t)l16 * 6400;

  FragC acc;
#pragma unroll
  for (int i = 0; i < 8; ++i) acc.f[i] = 0.0f;

  for (int kc = 0; kc < 6400; kc += 32) {
    Frag16 ahi, alo, bhi, blo;
#pragma unroll
    for (int v = 0; v < 8; ++v) {
      int kb = kc + ((v >> 2) << 4) + (halfL << 3) + ((v & 3) << 1);
      ahi.u[v] = *(const unsigned int*)(ah + kb);
      alo.u[v] = *(const unsigned int*)(al + kb);
      int kB = kc + ((v >> 2) << 4) + (halfL << 3) + ((v & 3) << 1);
      bhi.u[v] = *(const unsigned int*)(bh + kB);
      blo.u[v] = *(const unsigned int*)(bl + kB);
    }
    acc.v = __builtin_amdgcn_wmma_f32_16x16x32_bf16(false, ahi.v, false, bhi.v, (short)0, acc.v, false, false);
    acc.v = __builtin_amdgcn_wmma_f32_16x16x32_bf16(false, ahi.v, false, blo.v, (short)0, acc.v, false, false);
    acc.v = __builtin_amdgcn_wmma_f32_16x16x32_bf16(false, alo.v, false, bhi.v, (short)0, acc.v, false, false);
  }

  if (l16 < 10) {
    float bv = fcb[l16];
#pragma unroll
    for (int v = 0; v < 8; ++v)
      out[(mtile * 16 + halfL * 8 + v) * 10 + l16] = acc.f[v] + bv;
  }
}

// ---------- launch ----------
extern "C" void kernel_launch(void* const* d_in, const int* in_sizes, int n_in,
                              void* d_out, int out_size, void* d_ws, size_t ws_size,
                              hipStream_t stream) {
  const float* x   = (const float*)d_in[0];
  const float* W1  = (const float*)d_in[1];
  const float* b1  = (const float*)d_in[2];
  const float* g1  = (const float*)d_in[3];
  const float* be1 = (const float*)d_in[4];
  const float* W2  = (const float*)d_in[5];
  const float* b2  = (const float*)d_in[6];
  const float* g2  = (const float*)d_in[7];
  const float* be2 = (const float*)d_in[8];
  const float* W3  = (const float*)d_in[9];
  const float* b3  = (const float*)d_in[10];
  const float* g3  = (const float*)d_in[11];
  const float* be3 = (const float*)d_in[12];
  const float* fcW = (const float*)d_in[13];
  const float* fcb = (const float*)d_in[14];
  float* out = (float*)d_out;

  char* ws = (char*)d_ws;
  size_t off = 0;
  auto carve = [&](size_t bytes) {
    void* p = ws + off;
    off = (off + bytes + 255) & ~(size_t)255;
    return p;
  };
  unsigned int*   act0  = (unsigned int*)  carve(256u * BATCH * 4);          // (1,16,16,B) packed
  unsigned short* w1hi  = (unsigned short*)carve(196u * 16u * 32u * 2);      // (loc,O,Kpad)
  unsigned short* w1lo  = (unsigned short*)carve(196u * 16u * 32u * 2);
  float*          raw1  = (float*)         carve((size_t)16 * 196 * BATCH * 4);
  unsigned int*   act1  = (unsigned int*)  carve((size_t)16 * 196 * BATCH * 4);
  unsigned short* w2hi  = (unsigned short*)carve(144u * 32u * 160u * 2);
  unsigned short* w2lo  = (unsigned short*)carve(144u * 32u * 160u * 2);
  float*          raw2  = (float*)         carve((size_t)32 * 144 * BATCH * 4);
  unsigned int*   act2  = (unsigned int*)  carve((size_t)32 * 144 * BATCH * 4);
  unsigned short* w3hi  = (unsigned short*)carve(100u * 64u * 288u * 2);
  unsigned short* w3lo  = (unsigned short*)carve(100u * 64u * 288u * 2);
  float*          raw3  = (float*)         carve((size_t)64 * 100 * BATCH * 4);
  unsigned short* a3hi  = (unsigned short*)carve((size_t)BATCH * 6400 * 2);  // (B,6400)
  unsigned short* a3lo  = (unsigned short*)carve((size_t)BATCH * 6400 * 2);
  unsigned short* fwhi  = (unsigned short*)carve(16u * 6400u * 2);
  unsigned short* fwlo  = (unsigned short*)carve(16u * 6400u * 2);
  float*          stats = (float*)         carve(6u * 64u * 4);
  float *sum1 = stats, *sq1 = stats + 64, *sum2 = stats + 128, *sq2 = stats + 192,
        *sum3 = stats + 256, *sq3 = stats + 320;

  // prep
  zero_stats<<<2, 256, 0, stream>>>(stats, 384);
  pack_input<<<(256 * BATCH + 255) / 256, 256, 0, stream>>>(x, act0);
  pack_lcw<<<(196 * 16 * 32 + 255) / 256, 256, 0, stream>>>(W1, w1hi, w1lo, 16, 1, 196, 9, 32);
  pack_lcw<<<(144 * 32 * 160 + 255) / 256, 256, 0, stream>>>(W2, w2hi, w2lo, 32, 16, 144, 144, 160);
  pack_lcw<<<(100 * 64 * 288 + 255) / 256, 256, 0, stream>>>(W3, w3hi, w3lo, 64, 32, 100, 288, 288);
  pack_fcw<<<(16 * 6400 + 255) / 256, 256, 0, stream>>>(fcW, fwhi, fwlo);

  // layer 1: 196 x (M=1024, K=9->32, N=16)
  lc_gemm<1><<<dim3(196, BATCH / MBLOCK), 256, 32 * MBLOCK * 4, stream>>>(
      act0, w1hi, w1lo, b1, raw1, sum1, sq1, 1, 16, 16, 14, 14, 9, 32);
  bn_act<<<(16 * 196 * BATCH + 255) / 256, 256, 0, stream>>>(
      raw1, sum1, sq1, g1, be1, act1, 196, 16 * 196 * BATCH);

  // layer 2: 144 x (M=1024, K=144->160, N=32)
  lc_gemm<2><<<dim3(144, BATCH / MBLOCK), 256, 160 * MBLOCK * 4, stream>>>(
      act1, w2hi, w2lo, b2, raw2, sum2, sq2, 16, 14, 14, 12, 12, 144, 160);
  bn_act<<<(32 * 144 * BATCH + 255) / 256, 256, 0, stream>>>(
      raw2, sum2, sq2, g2, be2, act2, 144, 32 * 144 * BATCH);

  // layer 3: 100 x (M=1024, K=288, N=64), tanh -> FC planes
  lc_gemm<4><<<dim3(100, BATCH / MBLOCK), 256, 288 * MBLOCK * 4, stream>>>(
      act2, w3hi, w3lo, b3, raw3, sum3, sq3, 32, 12, 12, 10, 10, 288, 288);
  bn_act_fc<<<(64 * 100 * BATCH + 255) / 256, 256, 0, stream>>>(
      raw3, sum3, sq3, g3, be3, a3hi, a3lo);

  // FC: (1024,6400) @ (6400,10)
  fc_gemm<<<64, 32, 0, stream>>>(a3hi, a3lo, fwhi, fwlo, fcb, out);
}
